// CrossAttention_14010183320278
// MI455X (gfx1250) — matrix operand order repaired and checked by reading, not compiled, not run on hardware
//
#include <hip/hip_runtime.h>
#include <hip/hip_bf16.h>

typedef __attribute__((ext_vector_type(16))) _Float16 v16h;
typedef __attribute__((ext_vector_type(8)))  _Float16 v8h;
typedef __attribute__((ext_vector_type(8)))  float    v8f;
typedef __attribute__((ext_vector_type(4)))  int      v4i;

static __device__ __forceinline__ v16h cat16(v8h lo, v8h hi) {
  v16h r;
#pragma unroll
  for (int i = 0; i < 8; ++i) { r[i] = lo[i]; r[i + 8] = hi[i]; }
  return r;
}

static __device__ __forceinline__ v8f wmma_f16(v16h a, v16h b, v8f c) {
  // D = A(16x32 f16) * B(32x16 f16) + C(16x16 f32)
  return __builtin_amdgcn_wmma_f32_16x16x32_f16(false, a, false, b, (short)0, c,
                                                false, false);
}

// ---------------------------------------------------------------------------
// CDNA5 async global->LDS copy (GLOBAL_LOAD_ASYNC_TO_LDS_B128, ASYNCcnt).
// Builtin signature (from hipcc diagnostic): (as1 v4i*, as3 v4i*, imm, imm).
// Falls back to a plain VGPR round-trip copy if the builtin is unavailable.
// ---------------------------------------------------------------------------
#if defined(__gfx1250__) && __has_builtin(__builtin_amdgcn_global_load_async_to_lds_b128)
#define HAVE_ASYNC_LDS 1
#else
#define HAVE_ASYNC_LDS 0
#endif

typedef __attribute__((address_space(1))) v4i g_v4i;
typedef __attribute__((address_space(3))) v4i l_v4i;

static __device__ __forceinline__ void async_copy16(const _Float16* g,
                                                    _Float16* l) {
#if HAVE_ASYNC_LDS
  __builtin_amdgcn_global_load_async_to_lds_b128((g_v4i*)g, (l_v4i*)l, 0, 0);
#else
  *(v8h*)l = *(const v8h*)g;
#endif
}

template <int N>
static __device__ __forceinline__ void async_wait() {
#if HAVE_ASYNC_LDS
#if __has_builtin(__builtin_amdgcn_s_wait_asynccnt)
  __builtin_amdgcn_s_wait_asynccnt(N);
#else
  asm volatile("s_wait_asynccnt %0" ::"i"(N) : "memory");
#endif
#endif
}

// ---------------------------------------------------------------------------
// LayerNorm: fp32 [rows, C] -> f16 [rows, C]
// ---------------------------------------------------------------------------
template <int C>
__global__ __launch_bounds__(256) void ln_kernel(const float* __restrict__ x,
                                                 const float* __restrict__ gamma,
                                                 const float* __restrict__ beta,
                                                 _Float16* __restrict__ y) {
  const int row = blockIdx.x;
  const float* xr = x + (size_t)row * C;
  float vals[C / 256];
  float sum = 0.0f, sq = 0.0f;
#pragma unroll
  for (int i = 0; i < C / 256; ++i) {
    float v = xr[threadIdx.x + i * 256];
    vals[i] = v;
    sum += v;
    sq += v * v;
  }
#pragma unroll
  for (int m = 16; m >= 1; m >>= 1) {
    sum += __shfl_xor(sum, m, 32);
    sq += __shfl_xor(sq, m, 32);
  }
  __shared__ float ssum[8], ssq[8];
  const int wave = threadIdx.x >> 5;
  if ((threadIdx.x & 31) == 0) { ssum[wave] = sum; ssq[wave] = sq; }
  __syncthreads();
  sum = 0.0f; sq = 0.0f;
#pragma unroll
  for (int w = 0; w < 8; ++w) { sum += ssum[w]; sq += ssq[w]; }
  const float mu = sum / (float)C;
  const float var = sq / (float)C - mu * mu;
  const float inv = rsqrtf(var + 1e-5f);
  _Float16* yr = y + (size_t)row * C;
#pragma unroll
  for (int i = 0; i < C / 256; ++i) {
    const int c = threadIdx.x + i * 256;
    yr[c] = (_Float16)((vals[i] - mu) * inv * gamma[c] + beta[c]);
  }
}

// ---------------------------------------------------------------------------
// Weight transpose + f32->f16: W[K,N] -> Wt[N,K]   (one-time, LDS-tiled)
// ---------------------------------------------------------------------------
__global__ __launch_bounds__(256) void wt_kernel(const float* __restrict__ W,
                                                 _Float16* __restrict__ Wt,
                                                 int K, int N) {
  __shared__ _Float16 t[32][33];
  const int n0 = blockIdx.x * 32;
  const int k0 = blockIdx.y * 32;
  const int tx = threadIdx.x & 31, ty = threadIdx.x >> 5;
#pragma unroll
  for (int i = 0; i < 32; i += 8)
    t[ty + i][tx] = (_Float16)W[(size_t)(k0 + ty + i) * N + n0 + tx];
  __syncthreads();
#pragma unroll
  for (int i = 0; i < 32; i += 8)
    Wt[(size_t)(n0 + ty + i) * K + k0 + tx] = t[tx][ty + i];
}

// ---------------------------------------------------------------------------
// V transpose (f16): V[b*2048+key][1024] -> Vt[(b*1024+c)][2048], c = h*64+d
// ---------------------------------------------------------------------------
__global__ __launch_bounds__(256) void vt_kernel(const _Float16* __restrict__ V,
                                                 _Float16* __restrict__ Vt) {
  __shared__ _Float16 t[32][33];
  const int b = blockIdx.z;
  const int c0 = blockIdx.x * 32;  // channel
  const int r0 = blockIdx.y * 32;  // key
  const int tx = threadIdx.x & 31, ty = threadIdx.x >> 5;
#pragma unroll
  for (int i = 0; i < 32; i += 8)
    t[ty + i][tx] = V[(size_t)(b * 2048 + r0 + ty + i) * 1024 + c0 + tx];
  __syncthreads();
#pragma unroll
  for (int i = 0; i < 32; i += 8)
    Vt[(size_t)(b * 1024 + c0 + ty + i) * 2048 + r0 + tx] = t[tx][ty + i];
}

// ---------------------------------------------------------------------------
// GEMM: C[M,1024] = A[M,K](f16) @ Wt[1024,K](f16, pre-transposed) + bias(f32)
// Block tile 128x128, BK=32, 8 waves (4x2), wave tile 32x64 (2x4 WMMA frags).
// Double-buffered LDS with async global->LDS copies.
// ---------------------------------------------------------------------------
template <typename OutT, int K>
__global__ __launch_bounds__(256) void gemm_bias_kernel(
    const _Float16* __restrict__ A, const _Float16* __restrict__ Wt,
    const float* __restrict__ bias, OutT* __restrict__ C) {
  constexpr int N = 1024;
  __shared__ __align__(16) _Float16 lA[2][128][40];  // [m][k], 80B row stride
  __shared__ __align__(16) _Float16 lB[2][128][40];  // [n][k]

  const int tid = threadIdx.x;
  const int lane = tid & 31, wave = tid >> 5;
  const int lh = lane & 15, sel = lane >> 4;
  const int wm = (wave & 3) * 32, wn = (wave >> 2) * 64;
  const int mBlock = blockIdx.y * 128, nBlock = blockIdx.x * 128;

  v8f acc[2][4];
#pragma unroll
  for (int i = 0; i < 2; ++i)
#pragma unroll
    for (int j = 0; j < 4; ++j)
#pragma unroll
      for (int r = 0; r < 8; ++r) acc[i][j][r] = 0.0f;

  // Staging: each thread copies 2x16B of A and 2x16B of B per tile.
  const int ar = tid >> 1, ac = (tid & 1) * 16;
  const _Float16* gA = A + (size_t)(mBlock + ar) * K + ac;
  const _Float16* gB = Wt + (size_t)(nBlock + ar) * K + ac;

  auto stage = [&](int buf, int k0) {
    async_copy16(gA + k0, &lA[buf][ar][ac]);
    async_copy16(gA + k0 + 8, &lA[buf][ar][ac + 8]);
    async_copy16(gB + k0, &lB[buf][ar][ac]);
    async_copy16(gB + k0 + 8, &lB[buf][ar][ac + 8]);
  };
  auto compute = [&](int buf) {
    v16h af[2], bf[4];
#pragma unroll
    for (int mi = 0; mi < 2; ++mi) {
      const int m = wm + mi * 16 + lh;
      af[mi] = cat16(*(const v8h*)&lA[buf][m][sel * 8],
                     *(const v8h*)&lA[buf][m][16 + sel * 8]);
    }
#pragma unroll
    for (int ni = 0; ni < 4; ++ni) {
      const int n = wn + ni * 16 + lh;
      bf[ni] = cat16(*(const v8h*)&lB[buf][n][sel * 8],
                     *(const v8h*)&lB[buf][n][16 + sel * 8]);
    }
#pragma unroll
    for (int mi = 0; mi < 2; ++mi)
#pragma unroll
      for (int ni = 0; ni < 4; ++ni)
        acc[mi][ni] = wmma_f16(af[mi], bf[ni], acc[mi][ni]);
  };

  stage(0, 0);
  int buf = 0;
#pragma unroll 1
  for (int k0 = 0; k0 < K - 32; k0 += 32) {
    stage(buf ^ 1, k0 + 32);   // overlap next tile's copies with this compute
    async_wait<4>();           // only the previous tile's 4 ops must retire
    __syncthreads();
    compute(buf);
    __syncthreads();           // everyone done reading buf before restage
    buf ^= 1;
  }
  async_wait<0>();
  __syncthreads();
  compute(buf);

#pragma unroll
  for (int mi = 0; mi < 2; ++mi) {
#pragma unroll
    for (int ni = 0; ni < 4; ++ni) {
      const int gm0 = mBlock + wm + mi * 16 + sel * 8;
      const int gn = nBlock + wn + ni * 16 + lh;
      const float bv = bias[gn];
#pragma unroll
      for (int r = 0; r < 8; ++r)
        C[(size_t)(gm0 + r) * N + gn] = (OutT)(acc[mi][ni][r] + bv);
    }
  }
}

// ---------------------------------------------------------------------------
// Flash attention, one (b, h, 128-query tile) per block; D=64, K-tile=128.
// Q/K/O: f16 [B*2048, 1024], head h at cols h*64..+63.  Vt: f16 [(b*1024+c), 2048].
// ---------------------------------------------------------------------------
__global__ __launch_bounds__(256) void flash_attn_kernel(
    const _Float16* __restrict__ Q, const _Float16* __restrict__ Kp,
    const _Float16* __restrict__ Vt, _Float16* __restrict__ O) {
  // sm0 holds the K tile [128][72] during scores, then the P tiles [8][16][136]
  __shared__ __align__(16) _Float16 sm0[8 * 16 * 136];  // 34816 B
  __shared__ __align__(16) _Float16 smV[64 * 136];      // V^T tile, 17408 B
#define LK(r, c) sm0[(r) * 72 + (c)]
#define LP(w, r, c) sm0[(((w) * 16) + (r)) * 136 + (c)]
#define LVT(r, c) smV[(r) * 136 + (c)]

  const int tid = threadIdx.x;
  const int lane = tid & 31, wave = tid >> 5;
  const int lh = lane & 15, sel = lane >> 4;
  const int b = blockIdx.z, h = blockIdx.y;
  const int qBase = blockIdx.x * 128;
  const int hc = h * 64;

  // Q A-fragments (this wave owns query rows qBase + wave*16 .. +15)
  const _Float16* qrow =
      Q + (size_t)(b * 2048 + qBase + wave * 16 + lh) * 1024 + hc;
  v16h qf[2];
#pragma unroll
  for (int ks = 0; ks < 2; ++ks)
    qf[ks] = cat16(*(const v8h*)(qrow + ks * 32 + sel * 8),
                   *(const v8h*)(qrow + ks * 32 + 16 + sel * 8));

  float rowmax[8], rowsum[8];
  v8f oacc[4];
#pragma unroll
  for (int r = 0; r < 8; ++r) { rowmax[r] = -1e30f; rowsum[r] = 0.0f; }
#pragma unroll
  for (int d = 0; d < 4; ++d)
#pragma unroll
    for (int r = 0; r < 8; ++r) oacc[d][r] = 0.0f;

  const float sc = 0.125f * 1.44269504088896f;  // 1/sqrt(64) * log2(e)

  // K staging: thread -> (row sr, 32-half segment sc0); V^T staging: (vd, vc)
  const int sr = tid >> 1, sc0 = (tid & 1) * 32;
  const int vd = tid >> 2, vc = (tid & 3) * 32;
  const _Float16* gK0 = Kp + (size_t)(b * 2048 + sr) * 1024 + hc + sc0;
  const _Float16* gV0 = Vt + (size_t)(b * 1024 + hc + vd) * 2048 + vc;

  for (int kt = 0; kt < 2048 / 128; ++kt) {
    const int kBase = kt * 128;
    __syncthreads();  // previous iteration's P/V^T reads complete
    {
      const _Float16* gk = gK0 + (size_t)kBase * 1024;
#pragma unroll
      for (int j = 0; j < 4; ++j) async_copy16(gk + j * 8, &LK(sr, sc0 + j * 8));
      const _Float16* gv = gV0 + kBase;
#pragma unroll
      for (int j = 0; j < 4; ++j) async_copy16(gv + j * 8, &LVT(vd, vc + j * 8));
    }
    async_wait<0>();
    __syncthreads();

    // S = Q * K^T : K tile rows [key][d] are exactly the B-operand [n][k] layout
    v8f sacc[8];
#pragma unroll
    for (int ni = 0; ni < 8; ++ni)
#pragma unroll
      for (int r = 0; r < 8; ++r) sacc[ni][r] = 0.0f;
#pragma unroll
    for (int ks = 0; ks < 2; ++ks)
#pragma unroll
      for (int ni = 0; ni < 8; ++ni) {
        const int n = ni * 16 + lh;
        v16h bfr = cat16(*(const v8h*)&LK(n, ks * 32 + sel * 8),
                         *(const v8h*)&LK(n, ks * 32 + 16 + sel * 8));
        sacc[ni] = wmma_f16(qf[ks], bfr, sacc[ni]);
      }

    // online softmax (rows r+8*sel live in lanes (sel*16..sel*16+15))
    float corr[8];
#pragma unroll
    for (int r = 0; r < 8; ++r) {
      float m = sacc[0][r];
#pragma unroll
      for (int ni = 1; ni < 8; ++ni) m = fmaxf(m, sacc[ni][r]);
      m = fmaxf(m, __shfl_xor(m, 1, 32));
      m = fmaxf(m, __shfl_xor(m, 2, 32));
      m = fmaxf(m, __shfl_xor(m, 4, 32));
      m = fmaxf(m, __shfl_xor(m, 8, 32));
      const float nm = fmaxf(rowmax[r], m);
      corr[r] = exp2f((rowmax[r] - nm) * sc);
      rowmax[r] = nm;
    }
    float tsum[8];
#pragma unroll
    for (int r = 0; r < 8; ++r) tsum[r] = 0.0f;
#pragma unroll
    for (int ni = 0; ni < 8; ++ni)
#pragma unroll
      for (int r = 0; r < 8; ++r) {
        const float p = exp2f((sacc[ni][r] - rowmax[r]) * sc);
        sacc[ni][r] = p;
        tsum[r] += p;
      }
#pragma unroll
    for (int r = 0; r < 8; ++r) {
      float t = tsum[r];
      t += __shfl_xor(t, 1, 32);
      t += __shfl_xor(t, 2, 32);
      t += __shfl_xor(t, 4, 32);
      t += __shfl_xor(t, 8, 32);
      rowsum[r] = rowsum[r] * corr[r] + t;
    }
#pragma unroll
    for (int d = 0; d < 4; ++d)
#pragma unroll
      for (int r = 0; r < 8; ++r) oacc[d][r] *= corr[r];

    __syncthreads();  // all K-tile reads done; reuse sm0 for P
#pragma unroll
    for (int ni = 0; ni < 8; ++ni)
#pragma unroll
      for (int r = 0; r < 8; ++r)
        LP(wave, sel * 8 + r, ni * 16 + lh) = (_Float16)sacc[ni][r];
    __syncthreads();

    // O += P(16x128) * V(128x64)
#pragma unroll
    for (int ks2 = 0; ks2 < 4; ++ks2) {
      v16h pf = cat16(*(const v8h*)&LP(wave, lh, ks2 * 32 + sel * 8),
                      *(const v8h*)&LP(wave, lh, ks2 * 32 + 16 + sel * 8));
#pragma unroll
      for (int d = 0; d < 4; ++d) {
        const int n = d * 16 + lh;
        v16h bfr = cat16(*(const v8h*)&LVT(n, ks2 * 32 + sel * 8),
                         *(const v8h*)&LVT(n, ks2 * 32 + 16 + sel * 8));
        oacc[d] = wmma_f16(pf, bfr, oacc[d]);
      }
    }
  }

#pragma unroll
  for (int r = 0; r < 8; ++r) {
    const float inv = 1.0f / rowsum[r];
    const size_t row =
        (size_t)(b * 2048 + qBase + wave * 16 + sel * 8 + r) * 1024 + hc;
#pragma unroll
    for (int d = 0; d < 4; ++d)
      O[row + d * 16 + lh] = (_Float16)(oacc[d][r] * inv);
  }
#undef LK
#undef LP
#undef LVT
}

// ---------------------------------------------------------------------------
// Launch
// ---------------------------------------------------------------------------
extern "C" void kernel_launch(void* const* d_in, const int* in_sizes, int n_in,
                              void* d_out, int out_size, void* d_ws,
                              size_t ws_size, hipStream_t stream) {
  (void)in_sizes; (void)n_in; (void)out_size; (void)ws_size;
  const float* qtok = (const float*)d_in[0];
  const float* ctok = (const float*)d_in[1];
  const float* Wq = (const float*)d_in[2];
  const float* bq = (const float*)d_in[3];
  const float* Wk = (const float*)d_in[4];
  const float* bk = (const float*)d_in[5];
  const float* Wv = (const float*)d_in[6];
  const float* bv = (const float*)d_in[7];
  const float* Wo = (const float*)d_in[8];
  const float* bo = (const float*)d_in[9];
  const float* gamma_q = (const float*)d_in[10];
  const float* beta_q = (const float*)d_in[11];
  const float* gamma_c = (const float*)d_in[12];
  const float* beta_c = (const float*)d_in[13];

  size_t off = 0;
  auto carve = [&](size_t bytes) {
    void* p = (char*)d_ws + off;
    off += (bytes + 255) & ~(size_t)255;
    return p;
  };
  _Float16* xq16 = (_Float16*)carve(8192ull * 1024 * 2);
  _Float16* xc16 = (_Float16*)carve(8192ull * 768 * 2);
  _Float16* WqT = (_Float16*)carve(1024ull * 1024 * 2);
  _Float16* WkT = (_Float16*)carve(1024ull * 768 * 2);
  _Float16* WvT = (_Float16*)carve(1024ull * 768 * 2);
  _Float16* WoT = (_Float16*)carve(1024ull * 1024 * 2);
  _Float16* Q16 = (_Float16*)carve(8192ull * 1024 * 2);
  _Float16* K16 = (_Float16*)carve(8192ull * 1024 * 2);
  _Float16* V16 = (_Float16*)carve(8192ull * 1024 * 2);
  _Float16* Vt16 = (_Float16*)carve(4ull * 1024 * 2048 * 2);
  _Float16* A16 = xq16;  // xq16 dead after Q projection; reuse for attended

  // LayerNorms (fp32 -> f16)
  ln_kernel<1024><<<8192, 256, 0, stream>>>(qtok, gamma_q, beta_q, xq16);
  ln_kernel<768><<<8192, 256, 0, stream>>>(ctok, gamma_c, beta_c, xc16);

  // Weight transpose + f16 conversion: Wt[N,K]
  wt_kernel<<<dim3(32, 32), 256, 0, stream>>>(Wq, WqT, 1024, 1024);
  wt_kernel<<<dim3(32, 24), 256, 0, stream>>>(Wk, WkT, 768, 1024);
  wt_kernel<<<dim3(32, 24), 256, 0, stream>>>(Wv, WvT, 768, 1024);
  wt_kernel<<<dim3(32, 32), 256, 0, stream>>>(Wo, WoT, 1024, 1024);

  // Projections (WMMA GEMMs, double-buffered async staging)
  dim3 gq(1024 / 128, 8192 / 128);
  gemm_bias_kernel<_Float16, 1024><<<gq, 256, 0, stream>>>(xq16, WqT, bq, Q16);
  gemm_bias_kernel<_Float16, 768><<<gq, 256, 0, stream>>>(xc16, WkT, bk, K16);
  gemm_bias_kernel<_Float16, 768><<<gq, 256, 0, stream>>>(xc16, WvT, bv, V16);

  // Pre-transpose V per (b, head-channel) for the flash kernel's B operand
  vt_kernel<<<dim3(1024 / 32, 2048 / 32, 4), 256, 0, stream>>>(V16, Vt16);

  // Flash attention: grid (q-tiles, heads, batch)
  flash_attn_kernel<<<dim3(2048 / 128, 16, 4), 256, 0, stream>>>(Q16, K16, Vt16, A16);

  // Output projection (f32 out)
  gemm_bias_kernel<float, 1024><<<gq, 256, 0, stream>>>(A16, WoT, bo, (float*)d_out);
}
